// ForexLSTM_22548578304319
// MI455X (gfx1250) — matrix operand order repaired and checked by reading, not compiled, not run on hardware
//
#include <hip/hip_runtime.h>

// ForexLSTM on MI455X (gfx1250): 2-layer LSTM (H=8, IN=14) over T=512, B=4096,
// fused BN + MLP head. One wave32 owns a 16-row batch tile; the per-step
// matmuls are fused into 4x v_wmma_f32_16x16x32_f16 per timestep with bias in
// the C accumulator. D-layout -> A-layout transposes go through per-wave LDS
// with s_wait_dscnt ordering (same-wave, no barrier). The i/f <-> g/o lane
// swap uses DPP row_ror:8 (pure VALU). Sigmoid is tanh-based (1 TRANS op via
// native v_tanh_f32), and next-step x loads are software-pipelined behind the
// cell math to hide HBM/L2 latency.

#define HID   8
#define IN_F  14
#define B_TOT 4096
#define T_LEN 512

typedef __attribute__((ext_vector_type(16))) _Float16 v16h;
typedef __attribute__((ext_vector_type(8)))  _Float16 v8h;
typedef __attribute__((ext_vector_type(8)))  float    v8f;

__device__ __forceinline__ float tanh_f(float x) {
#if __has_builtin(__builtin_amdgcn_tanhf)
  return __builtin_amdgcn_tanhf(x);   // v_tanh_f32 on gfx1250
#else
  float ax = fabsf(x);
  float e  = __expf(2.0f * ax);                             // >= 1, may be +inf
  float t  = 1.0f - 2.0f * __builtin_amdgcn_rcpf(e + 1.0f); // inf -> rcp 0 -> 1
  return copysignf(t, x);
#endif
}

__device__ __forceinline__ float sigm_f(float x) {
  // sigmoid(x) = 0.5*tanh(0.5x) + 0.5 : single v_tanh_f32 + mul + fma
  return __builtin_fmaf(0.5f, tanh_f(0.5f * x), 0.5f);
}

// lane ^ 8 swap within each 16-lane row: DPP row_ror:8 (ctrl 0x128), no LDS.
__device__ __forceinline__ float xor8_dpp(float v) {
  return __int_as_float(__builtin_amdgcn_update_dpp(
      0, __float_as_int(v), 0x128, 0xf, 0xf, true));
}

__global__ __launch_bounds__(256) void forex_lstm_kernel(
    const float* __restrict__ x,
    const float* __restrict__ Wih1, const float* __restrict__ Whh1,
    const float* __restrict__ bih1, const float* __restrict__ bhh1,
    const float* __restrict__ Wih2, const float* __restrict__ Whh2,
    const float* __restrict__ bih2, const float* __restrict__ bhh2,
    const float* __restrict__ bn_g, const float* __restrict__ bn_b,
    const float* __restrict__ bn_m, const float* __restrict__ bn_v,
    const float* __restrict__ mw1, const float* __restrict__ mb1,
    const float* __restrict__ mw2, const float* __restrict__ mb2,
    float* __restrict__ out)
{
  // Per-wave LDS: [0,256) h1 f16 (16 rows x 8 halves), [256,512) h2 f16,
  //               [512,1024) final-h2 f32 stage for the epilogue.
  __shared__ __attribute__((aligned(16))) unsigned char lds_raw[8 * 1024];

  const int lane   = threadIdx.x & 31;
  const int wave   = threadIdx.x >> 5;
  const int tile   = blockIdx.x * 8 + wave;  // 16-row batch tile
  const int row0   = tile * 16;
  const int nlo    = lane & 15;              // A-row index M / gate column N%16
  const bool hiH   = lane >= 16;             // upper lane half
  const bool hi8   = (lane & 8) != 0;

  _Float16* h1buf = (_Float16*)(lds_raw + (size_t)wave * 1024);
  _Float16* h2buf = (_Float16*)(lds_raw + (size_t)wave * 1024 + 256);
  float*    epif  = (float*)   (lds_raw + (size_t)wave * 1024 + 512);

  // ---- zero-init h state staging (h1 + h2 = 512 B / wave) ----
  {
    unsigned int* z = (unsigned int*)h1buf;
#pragma unroll
    for (int i = 0; i < 4; ++i) z[lane * 4 + i] = 0u;
  }
  asm volatile("s_wait_dscnt 0" ::: "memory");

  // ---- build B fragments once (weights are tiny & loop-invariant) ----
  // 16-bit B 32x16: lane L holds column N=L%16; element e covers K = (e<8 ? e : e+8) + 8*(L>=16).
  v16h B1t0, B1t1, B2t0, B2t1;
  for (int e = 0; e < 16; ++e) {
    const int K = (e < 8 ? e : e + 8) + (hiH ? 8 : 0);
    float a0 = 0.f, a1 = 0.f, b0 = 0.f, b1 = 0.f;
    if (K < IN_F) {                       // layer-1 input weights
      a0 = Wih1[nlo * IN_F + K];
      a1 = Wih1[(nlo + 16) * IN_F + K];
    } else if (K >= 16 && K < 16 + HID) { // layer-1 recurrent weights
      a0 = Whh1[nlo * HID + (K - 16)];
      a1 = Whh1[(nlo + 16) * HID + (K - 16)];
    }
    if (K < HID) {                        // layer-2 input (=h1) weights
      b0 = Wih2[nlo * HID + K];
      b1 = Wih2[(nlo + 16) * HID + K];
    } else if (K < 2 * HID) {             // layer-2 recurrent weights
      b0 = Whh2[nlo * HID + (K - HID)];
      b1 = Whh2[(nlo + 16) * HID + (K - HID)];
    }
    B1t0[e] = (_Float16)a0; B1t1[e] = (_Float16)a1;
    B2t0[e] = (_Float16)b0; B2t1[e] = (_Float16)b1;
  }

  // ---- bias-preloaded C accumulators (C[m,n] = bias[n], all m) ----
  const float bs1t0 = bih1[nlo]      + bhh1[nlo];
  const float bs1t1 = bih1[nlo + 16] + bhh1[nlo + 16];
  const float bs2t0 = bih2[nlo]      + bhh2[nlo];
  const float bs2t1 = bih2[nlo + 16] + bhh2[nlo + 16];
  v8f Cb1t0, Cb1t1, Cb2t0, Cb2t1;
#pragma unroll
  for (int v = 0; v < 8; ++v) {
    Cb1t0[v] = bs1t0; Cb1t1[v] = bs1t1;
    Cb2t0[v] = bs2t0; Cb2t1[v] = bs2t1;
  }

  // ---- LSTM cell state in registers (unit-lane layout, replicated u <-> u+8) ----
  float c1[8], c2[8], h2f[8];
#pragma unroll
  for (int v = 0; v < 8; ++v) { c1[v] = 0.f; c2[v] = 0.f; h2f[v] = 0.f; }

  const float* xrow = x + (size_t)(row0 + nlo) * (T_LEN * IN_F);
  const unsigned hmask = hiH ? 0u : 0xffffffffu;  // zero h1-slice in hi lanes

  auto load_x = [&](int t, float xv[8]) {
    const float* xp = xrow + t * IN_F + (hiH ? 8 : 0);
#pragma unroll
    for (int i = 0; i < 6; ++i) xv[i] = xp[i];
    xv[6] = 0.f; xv[7] = 0.f;
    if (!hiH) { xv[6] = xp[6]; xv[7] = xp[7]; }
  };

  float xv[8];
  load_x(0, xv);   // prologue load for t = 0

  for (int t = 0; t < T_LEN; ++t) {
    // ---- A1 = [x_t | pad | h1 | pad] (16-bit A 16x32 layout) ----
    v16h A1;
#pragma unroll
    for (int i = 0; i < 8; ++i) A1[i] = (_Float16)xv[i];
    union { v8h h; unsigned u[4]; } hr;
    hr.h = *(const v8h*)(h1buf + nlo * 8);
#pragma unroll
    for (int i = 0; i < 4; ++i) hr.u[i] &= hmask;   // 4x v_and_b32
#pragma unroll
    for (int i = 0; i < 8; ++i) A1[8 + i] = hr.h[i];

    // ---- layer-1 gates: two 16x16x32 WMMAs (N-tiles {i|f} and {g|o}) ----
    v8f d0 = __builtin_amdgcn_wmma_f32_16x16x32_f16(false, A1, false, B1t0,
                                                    (short)0, Cb1t0, false, false);
    v8f d1 = __builtin_amdgcn_wmma_f32_16x16x32_f16(false, A1, false, B1t1,
                                                    (short)0, Cb1t1, false, false);

    // ---- software-pipelined x load for t+1 (clamped; dead in last iter) ----
    float xn[8];
    load_x(t + 1 < T_LEN ? t + 1 : t, xn);

    // ---- layer-1 cell (lanes u and u^8 hold i/f, g/o of the same unit) ----
    _Float16 h1h[8];
#pragma unroll
    for (int v = 0; v < 8; ++v) {
      float a0 = d0[v], a1 = d1[v];
      float p0 = xor8_dpp(a0);
      float p1 = xor8_dpp(a1);
      float gi = hi8 ? p0 : a0;
      float gf = hi8 ? a0 : p0;
      float gg = hi8 ? p1 : a1;
      float go = hi8 ? a1 : p1;
      float cn = sigm_f(gf) * c1[v] + sigm_f(gi) * tanh_f(gg);
      float hn = sigm_f(go) * tanh_f(cn);
      c1[v] = cn;
      h1h[v] = (_Float16)hn;
    }

    // ---- stage h1 (D-layout -> row-major f16) for the A-side transpose ----
    if (!hi8) {
      const int u = lane & 7;
      const int rb = hiH ? 8 : 0;
#pragma unroll
      for (int v = 0; v < 8; ++v) h1buf[(rb + v) * 8 + u] = h1h[v];
    }
    asm volatile("s_wait_dscnt 0" ::: "memory");

    // ---- A2 = [h1 | h2 | 0] : lo lanes read h1 rows, hi lanes h2 rows ----
    v16h A2;
    const _Float16* hb = hiH ? h2buf : h1buf;
    v8h hrow = *(const v8h*)(hb + nlo * 8);
#pragma unroll
    for (int i = 0; i < 8; ++i) { A2[i] = hrow[i]; A2[8 + i] = (_Float16)0; }

    v8f e0 = __builtin_amdgcn_wmma_f32_16x16x32_f16(false, A2, false, B2t0,
                                                    (short)0, Cb2t0, false, false);
    v8f e1 = __builtin_amdgcn_wmma_f32_16x16x32_f16(false, A2, false, B2t1,
                                                    (short)0, Cb2t1, false, false);

    // ---- layer-2 cell ----
    _Float16 h2h[8];
#pragma unroll
    for (int v = 0; v < 8; ++v) {
      float a0 = e0[v], a1 = e1[v];
      float p0 = xor8_dpp(a0);
      float p1 = xor8_dpp(a1);
      float gi = hi8 ? p0 : a0;
      float gf = hi8 ? a0 : p0;
      float gg = hi8 ? p1 : a1;
      float go = hi8 ? a1 : p1;
      float cn = sigm_f(gf) * c2[v] + sigm_f(gi) * tanh_f(gg);
      float hn = sigm_f(go) * tanh_f(cn);
      c2[v] = cn;
      h2f[v] = hn;
      h2h[v] = (_Float16)hn;
    }

    if (!hi8) {
      const int u = lane & 7;
      const int rb = hiH ? 8 : 0;
#pragma unroll
      for (int v = 0; v < 8; ++v) h2buf[(rb + v) * 8 + u] = h2h[v];
    }
    asm volatile("s_wait_dscnt 0" ::: "memory");

    // rotate pipelined x
#pragma unroll
    for (int i = 0; i < 8; ++i) xv[i] = xn[i];
  }

  // ---- epilogue: BatchNorm (eval) + ReLU MLP (8 -> 4 -> 1), f32 precision ----
  if (!hi8) {
    const int u = lane & 7;
    const int rb = hiH ? 8 : 0;
#pragma unroll
    for (int v = 0; v < 8; ++v) epif[(rb + v) * 8 + u] = h2f[v];
  }
  asm volatile("s_wait_dscnt 0" ::: "memory");

  if (lane < 16) {
    const int r = lane;
    float acc[4];
#pragma unroll
    for (int j = 0; j < 4; ++j) acc[j] = mb1[j];
#pragma unroll
    for (int u = 0; u < HID; ++u) {
      float hv = epif[r * 8 + u];
      float nv = bn_g[u] * (hv - bn_m[u]) * rsqrtf(bn_v[u] + 1e-5f) + bn_b[u];
#pragma unroll
      for (int j = 0; j < 4; ++j) acc[j] += nv * mw1[j * HID + u];
    }
    float o = mb2[0];
#pragma unroll
    for (int j = 0; j < 4; ++j) o += fmaxf(acc[j], 0.f) * mw2[j];
    out[row0 + r] = o;
  }
}

extern "C" void kernel_launch(void* const* d_in, const int* in_sizes, int n_in,
                              void* d_out, int out_size, void* d_ws, size_t ws_size,
                              hipStream_t stream) {
  (void)in_sizes; (void)n_in; (void)d_ws; (void)ws_size; (void)out_size;
  const float* x    = (const float*)d_in[0];
  const float* Wih1 = (const float*)d_in[1];
  const float* Whh1 = (const float*)d_in[2];
  const float* bih1 = (const float*)d_in[3];
  const float* bhh1 = (const float*)d_in[4];
  const float* Wih2 = (const float*)d_in[5];
  const float* Whh2 = (const float*)d_in[6];
  const float* bih2 = (const float*)d_in[7];
  const float* bhh2 = (const float*)d_in[8];
  const float* bn_g = (const float*)d_in[9];
  const float* bn_b = (const float*)d_in[10];
  const float* bn_m = (const float*)d_in[11];
  const float* bn_v = (const float*)d_in[12];
  const float* mw1  = (const float*)d_in[13];
  const float* mb1  = (const float*)d_in[14];
  const float* mw2  = (const float*)d_in[15];
  const float* mb2  = (const float*)d_in[16];
  float* out = (float*)d_out;

  // 16 batch rows per wave, 8 waves per block -> 128 rows/block, 32 blocks.
  forex_lstm_kernel<<<dim3(B_TOT / 128), dim3(256), 0, stream>>>(
      x, Wih1, Whh1, bih1, bhh1, Wih2, Whh2, bih2, bhh2,
      bn_g, bn_b, bn_m, bn_v, mw1, mb1, mw2, mb2, out);
}
